// CausalSelfAttention_19894288515671
// MI455X (gfx1250) — matrix operand order repaired and checked by reading, not compiled, not run on hardware
//
#include <hip/hip_runtime.h>
#include <hip/hip_bf16.h>

// CausalSelfAttention for MI455X (gfx1250): bf16 WMMA pipeline.
//  x[B,T,C] f32 -> bf16; q/k/v = x @ w^T via v_wmma_f32_16x16x32_bf16 with
//  TDM (tensor_load_to_lds) double-buffered tile staging;
//  RoPE applied twice == single rotation by 2*theta (precomputed tables);
//  flash-attention with online softmax (wave-per-16-query-rows);
//  y @ wo^T -> f32 d_out.

#define N_HEAD   16
#define HEAD_DIM 128
#define N_EMBD   2048
#define BATCH    4
#define SEQ      2048

typedef __bf16 bf16;
typedef __attribute__((ext_vector_type(16))) __bf16 v16bf;
typedef __attribute__((ext_vector_type(8)))  __bf16 v8bf;
typedef __attribute__((ext_vector_type(8)))  float  v8f;
typedef __attribute__((ext_vector_type(4)))  unsigned int u32x4;
typedef __attribute__((ext_vector_type(8)))  unsigned int u32x8;

// LDS tile geometry for TDM-staged GEMM operands: 32 bf16 (64B) per row,
// TDM pads +16B after every 64B (pad_interval code 3 = 16 dwords,
// pad_amount code 3 = 4 dwords) -> 80B row stride = 40 elements.
// 80B stride => rows 0..15 hit 16 distinct bank groups (20*r mod 64 distinct),
// and every ds_load_b128 stays 16B-aligned.
#define LDS_ROW_ELEMS 40
#define LDS_TILE_BYTES (128 * 80)

__device__ __forceinline__ int lane_id() { return (int)(threadIdx.x & 31u); }

// A-fragment (16x32 bf16), ISA 7.12.2: lane L -> row L%16,
// K-chunks [(L/16)*8 .. +7] (VGPR0-3) and [(L/16)*8+16 .. +23] (VGPR4-7).
__device__ __forceinline__ v16bf load_fragA(const bf16* p, int ld) {
  int l = lane_id();
  const bf16* q = p + (size_t)(l & 15) * (size_t)ld + ((l >> 4) << 3);
  v8bf lo = *(const v8bf*)q;
  v8bf hi = *(const v8bf*)(q + 16);
  return __builtin_shufflevector(lo, hi, 0,1,2,3,4,5,6,7,8,9,10,11,12,13,14,15);
}

// B-fragment (32x16 bf16) loaded from row-major B^T ([N,K]): lane L -> output
// column L%16 (= row L%16 of B^T), K-chunks [(L/16)*8 .. +7] and [(L/16)*8+16 .. +23].
__device__ __forceinline__ v16bf load_fragB(const bf16* p, int ld) {
  int l = lane_id();
  const bf16* q = p + (size_t)(l & 15) * (size_t)ld + ((l >> 4) << 3);
  v8bf lo = *(const v8bf*)q;
  v8bf hi = *(const v8bf*)(q + 16);
  return __builtin_shufflevector(lo, hi, 0,1,2,3,4,5,6,7,8,9,10,11,12,13,14,15);
}

__device__ __forceinline__ v8f wmma_bf16(v16bf a, v16bf b, v8f c) {
  // (neg_a, A, neg_b, B, c_mod, C, reuse_a, reuse_b)
  return __builtin_amdgcn_wmma_f32_16x16x32_bf16(false, a, false, b, (short)0, c,
                                                 false, false);
}

// ----------------------------------------------------------------- TDM -----
// Issue a TDM DMA of a [tile_d1 x 32] bf16 tile (row stride `stride0` elems in
// global) into LDS at byte offset lds_off, padding LDS rows to 80B.
// D# bitfields per CDNA5 ISA. Wave-level op
// (EXEC ignored); call from exactly one wave per block. Tracked by TENSORcnt.
__device__ __forceinline__ void tdm_load_tile32(const bf16* gsrc, unsigned lds_off,
                                                unsigned tensor_d0, unsigned tensor_d1,
                                                unsigned tile_d1,
                                                unsigned long long stride0) {
  unsigned long long ga = (unsigned long long)(size_t)gsrc;
  u32x4 g0;
  g0[0] = 1u;                                            // count=1, user D#
  g0[1] = lds_off;                                       // lds_addr (bytes)
  g0[2] = (unsigned)(ga & 0xFFFFFFFFu);                  // global_addr[31:0]
  g0[3] = (unsigned)((ga >> 32) & 0x01FFFFFFu) | (2u << 30);  // [56:32] | type=2
  u32x8 g1;
  g1[0] = (1u << 16)      // data_size = 1 (2 bytes)
        | (1u << 20)      // pad_enable
        | (3u << 22)      // pad_interval code 3 = 16 dwords (64B)
        | (3u << 25);     // pad_amount  code 3 = 4 dwords (16B)
  g1[1] = (tensor_d0 & 0xFFFFu) << 16;                            // dim0[15:0]
  g1[2] = ((tensor_d0 >> 16) & 0xFFFFu) | ((tensor_d1 & 0xFFFFu) << 16);
  g1[3] = ((tensor_d1 >> 16) & 0xFFFFu) | (32u << 16);            // tile_dim0=32
  g1[4] = tile_d1 & 0xFFFFu;                                      // tile_dim1
  g1[5] = (unsigned)(stride0 & 0xFFFFFFFFu);                      // stride0[31:0]
  g1[6] = (unsigned)((stride0 >> 32) & 0xFFFFu);                  // stride0[47:32]
  g1[7] = 0u;
  asm volatile("tensor_load_to_lds %0, %1" : : "s"(g0), "s"(g1) : "memory");
}

// ---------------------------------------------------------------- converts --
__global__ __launch_bounds__(256) void f32_to_bf16_k(const float* __restrict__ in,
                                                     bf16* __restrict__ out,
                                                     unsigned long long n) {
  unsigned long long i = (unsigned long long)blockIdx.x * blockDim.x + threadIdx.x;
  if (i < n) out[i] = (bf16)in[i];
}

// cos/sin of the DOUBLE angle (rotary applied twice in the reference).
__global__ __launch_bounds__(256) void rope_tables_k(float* __restrict__ c2,
                                                     float* __restrict__ s2) {
  int idx = blockIdx.x * blockDim.x + threadIdx.x;       // SEQ*64
  if (idx >= SEQ * 64) return;
  int t = idx >> 6, i = idx & 63;
  float inv = expf(-(float)i * (9.210340371976184f / 64.0f));  // 10000^(-i/64)
  float th  = 2.0f * (float)t * inv;
  c2[idx] = cosf(th);
  s2[idx] = sinf(th);
}

// ------------------------------------------------------------------- GEMM ---
// C[M,N] = A[M,K] (row-major bf16) * Bw[N,K]^T (row-major bf16).
// Block 256 thr = 8 waves (2x4); wave tile 64x32 => block tile 128x128.
// TDM double-buffers the 128x32 A and B tiles in LDS; wave 0 drives the DMA.
template <bool OUT_F32>
__global__ __launch_bounds__(256) void gemm_bt_k(const bf16* __restrict__ A,
                                                 const bf16* __restrict__ Bw,
                                                 float* __restrict__ Cf,
                                                 bf16* __restrict__ Cb,
                                                 int M, int N, int K) {
  __shared__ __align__(16) bf16 Ash[2][128 * LDS_ROW_ELEMS];   // 2 x 10KB
  __shared__ __align__(16) bf16 Bsh[2][128 * LDS_ROW_ELEMS];   // 2 x 10KB
  int warp = (int)(threadIdx.x >> 5);
  int wm = warp >> 2, wn = warp & 3;
  int row0 = blockIdx.y * 128;
  int col0 = blockIdx.x * 128;
  // Generic->LDS offset: LDS aperture addresses carry the byte offset in [31:0].
  unsigned aoff = (unsigned)(size_t)(&Ash[0][0]);
  unsigned boff = (unsigned)(size_t)(&Bsh[0][0]);

  v8f acc[4][2];
#pragma unroll
  for (int i = 0; i < 4; ++i)
#pragma unroll
    for (int j = 0; j < 2; ++j) acc[i][j] = (v8f)0.0f;

  int nk = K / 32;
  if (warp == 0) {
    tdm_load_tile32(A  + (size_t)row0 * K, aoff, (unsigned)K, (unsigned)M, 128,
                    (unsigned long long)K);
    tdm_load_tile32(Bw + (size_t)col0 * K, boff, (unsigned)K, (unsigned)N, 128,
                    (unsigned long long)K);
  }

  for (int it = 0; it < nk; ++it) {
    int cur = it & 1;
    __syncthreads();   // all waves done reading buf[cur^1] from the prev iter
    if (warp == 0) {
      if (it + 1 < nk) {
        int kb = (it + 1) * 32;
        tdm_load_tile32(A  + (size_t)row0 * K + kb,
                        aoff + (unsigned)(cur ^ 1) * LDS_TILE_BYTES,
                        (unsigned)K, (unsigned)M, 128, (unsigned long long)K);
        tdm_load_tile32(Bw + (size_t)col0 * K + kb,
                        boff + (unsigned)(cur ^ 1) * LDS_TILE_BYTES,
                        (unsigned)K, (unsigned)N, 128, (unsigned long long)K);
        // TDM completes in order: <=2 outstanding => current pair has landed.
        __builtin_amdgcn_s_wait_tensorcnt(2);
      } else {
        __builtin_amdgcn_s_wait_tensorcnt(0);
      }
    }
    __syncthreads();   // buf[cur] visible to all waves; next DMA overlaps math

    v16bf a[4], b[2];
#pragma unroll
    for (int i = 0; i < 4; ++i)
      a[i] = load_fragA(&Ash[cur][(wm * 64 + 16 * i) * LDS_ROW_ELEMS], LDS_ROW_ELEMS);
#pragma unroll
    for (int j = 0; j < 2; ++j)
      b[j] = load_fragB(&Bsh[cur][(wn * 32 + 16 * j) * LDS_ROW_ELEMS], LDS_ROW_ELEMS);
#pragma unroll
    for (int i = 0; i < 4; ++i)
#pragma unroll
      for (int j = 0; j < 2; ++j) acc[i][j] = wmma_bf16(a[i], b[j], acc[i][j]);
  }

  int l = lane_id();
  int cl = l & 15, rh = (l >> 4) << 3;       // C-tile: row=g+rh, col=cl
#pragma unroll
  for (int i = 0; i < 4; ++i)
#pragma unroll
    for (int j = 0; j < 2; ++j)
#pragma unroll
      for (int g = 0; g < 8; ++g) {
        size_t r = (size_t)(row0 + wm * 64 + 16 * i + rh + g);
        size_t c = (size_t)(col0 + wn * 32 + 16 * j + cl);
        if (OUT_F32) Cf[r * N + c] = acc[i][j][g];
        else         Cb[r * N + c] = (bf16)acc[i][j][g];
      }
}

// ---------------------------------------------------- RoPE + head-relayout --
// in:  [B,T,H*D] bf16 (GEMM output)   out: [B,H,T,D] bf16, rotated by 2*theta.
__global__ __launch_bounds__(256) void rope_regroup_k(const bf16* __restrict__ in,
                                                      bf16* __restrict__ out,
                                                      const float* __restrict__ c2,
                                                      const float* __restrict__ s2) {
  int idx = blockIdx.x * blockDim.x + threadIdx.x;   // B*T*H*64
  if (idx >= BATCH * SEQ * N_HEAD * 64) return;
  int i = idx & 63;
  int h = (idx >> 6) & 15;
  int t = (idx >> 10) & (SEQ - 1);
  int b = idx >> 21;
  size_t src = ((size_t)b * SEQ + t) * N_EMBD + (size_t)h * HEAD_DIM;
  float x1 = (float)in[src + i];
  float x2 = (float)in[src + 64 + i];
  float c = c2[t * 64 + i], s = s2[t * 64 + i];
  float y1 =  x1 * c + x2 * s;
  float y2 = -x1 * s + x2 * c;
  size_t dst = (((size_t)b * N_HEAD + h) * SEQ + t) * HEAD_DIM;
  out[dst + i]      = (bf16)y1;
  out[dst + 64 + i] = (bf16)y2;
}

// V: [B,T,H*D] -> [B,H,D,T] so PV B-fragments load contiguous keys.
__global__ __launch_bounds__(256) void vtrans_k(const bf16* __restrict__ in,
                                                bf16* __restrict__ out) {
  int idx = blockIdx.x * blockDim.x + threadIdx.x;   // B*H*D*T
  if (idx >= BATCH * N_HEAD * HEAD_DIM * SEQ) return;
  int t = idx & (SEQ - 1);
  int d = (idx >> 11) & (HEAD_DIM - 1);
  int h = (idx >> 18) & (N_HEAD - 1);
  int b = idx >> 22;
  out[idx] = in[((size_t)b * SEQ + t) * N_EMBD + (size_t)h * HEAD_DIM + d];
}

// -------------------------------------------------------- flash attention ---
// grid (SEQ/64, B*H); block 128 = 4 waves; each wave: 16 query rows x D=128.
// Per-wave causal trip counts -> no block barriers; fragments direct from L2
// (K/V for one (b,h) is 512KB, resident in the 192MB L2 across query blocks).
__global__ __launch_bounds__(128) void flash_attn_k(const bf16* __restrict__ qh,
                                                    const bf16* __restrict__ kh,
                                                    const bf16* __restrict__ vt,
                                                    bf16* __restrict__ yb) {
  __shared__ __align__(32) bf16 plds[4][16 * 32];   // per-wave P staging (1KB)
  int warp = (int)(threadIdx.x >> 5);
  int bh = blockIdx.y;
  int b = bh >> 4, h = bh & 15;
  int m0 = blockIdx.x * 64 + warp * 16;

  const bf16* qp = qh + (size_t)bh * SEQ * HEAD_DIM;
  const bf16* kp = kh + (size_t)bh * SEQ * HEAD_DIM;
  const bf16* vp = vt + (size_t)bh * HEAD_DIM * SEQ;

  v16bf qf[4];
#pragma unroll
  for (int dk = 0; dk < 4; ++dk)
    qf[dk] = load_fragA(qp + (size_t)m0 * HEAD_DIM + 32 * dk, HEAD_DIM);

  v8f acc[8];
#pragma unroll
  for (int t = 0; t < 8; ++t) acc[t] = (v8f)0.0f;
  float mrow[8], lrow[8];
#pragma unroll
  for (int g = 0; g < 8; ++g) { mrow[g] = -1.0e30f; lrow[g] = 0.0f; }

  int l = lane_id();
  int cl = l & 15, rh = (l >> 4) << 3;               // C-tile row=g+rh, col=cl
  const float scale = 0.08838834764831845f;          // 1/sqrt(128)
  bf16* pl = plds[warp];
  int kend = m0 + 16;                                // keys [0, m0+15]

  for (int k0 = 0; k0 < kend; k0 += 32) {
    // ---- scores S[16 x 32] = Q * K^T (8 WMMAs over D=128) ----
    v8f s[2];
    s[0] = (v8f)0.0f; s[1] = (v8f)0.0f;
#pragma unroll
    for (int n = 0; n < 2; ++n)
#pragma unroll
      for (int dk = 0; dk < 4; ++dk)
        s[n] = wmma_bf16(qf[dk],
                         load_fragB(kp + (size_t)(k0 + 16 * n) * HEAD_DIM + 32 * dk,
                                    HEAD_DIM),
                         s[n]);
    // scale + causal mask
#pragma unroll
    for (int n = 0; n < 2; ++n)
#pragma unroll
      for (int g = 0; g < 8; ++g) {
        int row = m0 + rh + g, col = k0 + 16 * n + cl;
        s[n][g] = s[n][g] * scale + (col > row ? -1.0e30f : 0.0f);
      }
    // row max across the 16-lane group
    float rmax[8];
#pragma unroll
    for (int g = 0; g < 8; ++g) rmax[g] = fmaxf(s[0][g], s[1][g]);
#pragma unroll
    for (int off = 1; off < 16; off <<= 1)
#pragma unroll
      for (int g = 0; g < 8; ++g)
        rmax[g] = fmaxf(rmax[g], __shfl_xor(rmax[g], off, 32));
    // online softmax update
    float alpha[8], rsum[8];
#pragma unroll
    for (int g = 0; g < 8; ++g) {
      float mn = fmaxf(mrow[g], rmax[g]);
      alpha[g] = __expf(mrow[g] - mn);
      mrow[g] = mn;
      float p0 = __expf(s[0][g] - mn);
      float p1 = __expf(s[1][g] - mn);
      s[0][g] = p0; s[1][g] = p1;
      rsum[g] = p0 + p1;
    }
#pragma unroll
    for (int off = 1; off < 16; off <<= 1)
#pragma unroll
      for (int g = 0; g < 8; ++g) rsum[g] += __shfl_xor(rsum[g], off, 32);
#pragma unroll
    for (int g = 0; g < 8; ++g) lrow[g] = lrow[g] * alpha[g] + rsum[g];
#pragma unroll
    for (int t = 0; t < 8; ++t)
#pragma unroll
      for (int g = 0; g < 8; ++g) acc[t][g] *= alpha[g];

    // ---- P: C-layout -> A-layout via per-wave LDS slice (in-order DS) ----
#pragma unroll
    for (int n = 0; n < 2; ++n)
#pragma unroll
      for (int g = 0; g < 8; ++g)
        pl[(rh + g) * 32 + 16 * n + cl] = (bf16)s[n][g];
    asm volatile("s_wait_dscnt 0x0" ::: "memory");   // wave-local LDS ordering
    v16bf pa = load_fragA(pl, 32);

    // ---- acc += P * V (8 WMMAs over D=128) ----
#pragma unroll
    for (int t = 0; t < 8; ++t)
      acc[t] = wmma_bf16(pa,
                         load_fragB(vp + (size_t)(16 * t) * SEQ + k0, SEQ),
                         acc[t]);
  }

  // normalize + store y as [B,T,H*D] bf16
#pragma unroll
  for (int g = 0; g < 8; ++g) mrow[g] = 1.0f / lrow[g];  // reuse as inv-l
#pragma unroll
  for (int t = 0; t < 8; ++t)
#pragma unroll
    for (int g = 0; g < 8; ++g) {
      size_t r = (size_t)(m0 + rh + g);
      size_t c = (size_t)(16 * t + cl);
      yb[(((size_t)b * SEQ + r) * N_HEAD + h) * HEAD_DIM + c] =
          (bf16)(acc[t][g] * mrow[g]);
    }
}

// ------------------------------------------------------------------ launch --
extern "C" void kernel_launch(void* const* d_in, const int* in_sizes, int n_in,
                              void* d_out, int out_size, void* d_ws, size_t ws_size,
                              hipStream_t stream) {
  const float* x  = (const float*)d_in[0];
  const float* wq = (const float*)d_in[1];
  const float* wk = (const float*)d_in[2];
  const float* wv = (const float*)d_in[3];
  const float* wo = (const float*)d_in[4];
  float* out = (float*)d_out;

  const size_t nX = (size_t)BATCH * SEQ * N_EMBD;       // 16,777,216
  const size_t nW = (size_t)N_EMBD * N_EMBD;            //  4,194,304
  const int    M  = BATCH * SEQ;                        //  8192

  char* w = (char*)d_ws;
  auto carve = [&](size_t bytes) {
    void* p = (void*)w;
    w += (bytes + 255) & ~(size_t)255;
    return p;
  };
  bf16* xb   = (bf16*)carve(nX * 2);
  bf16* wqb  = (bf16*)carve(nW * 2);
  bf16* wkb  = (bf16*)carve(nW * 2);
  bf16* wvb  = (bf16*)carve(nW * 2);
  bf16* wob  = (bf16*)carve(nW * 2);
  bf16* qtmp = (bf16*)carve(nX * 2);
  bf16* ktmp = (bf16*)carve(nX * 2);
  bf16* vtmp = (bf16*)carve(nX * 2);
  bf16* qhb  = (bf16*)carve(nX * 2);
  bf16* khb  = (bf16*)carve(nX * 2);
  bf16* vtb  = (bf16*)carve(nX * 2);
  float* c2  = (float*)carve((size_t)SEQ * 64 * 4);
  float* s2  = (float*)carve((size_t)SEQ * 64 * 4);
  bf16* yb   = qtmp;  // qtmp is dead after rope_regroup; reuse for y

  // 1) fp32 -> bf16
  f32_to_bf16_k<<<(unsigned)((nX + 255) / 256), 256, 0, stream>>>(x, xb, nX);
  f32_to_bf16_k<<<(unsigned)((nW + 255) / 256), 256, 0, stream>>>(wq, wqb, nW);
  f32_to_bf16_k<<<(unsigned)((nW + 255) / 256), 256, 0, stream>>>(wk, wkb, nW);
  f32_to_bf16_k<<<(unsigned)((nW + 255) / 256), 256, 0, stream>>>(wv, wvb, nW);
  f32_to_bf16_k<<<(unsigned)((nW + 255) / 256), 256, 0, stream>>>(wo, wob, nW);

  // 2) double-angle rotary tables
  rope_tables_k<<<(SEQ * 64 + 255) / 256, 256, 0, stream>>>(c2, s2);

  // 3) Q/K/V projections (TDM-staged WMMA GEMM, bf16 out)
  dim3 ggrid(N_EMBD / 128, M / 128);
  gemm_bt_k<false><<<ggrid, 256, 0, stream>>>(xb, wqb, nullptr, qtmp, M, N_EMBD, N_EMBD);
  gemm_bt_k<false><<<ggrid, 256, 0, stream>>>(xb, wkb, nullptr, ktmp, M, N_EMBD, N_EMBD);
  gemm_bt_k<false><<<ggrid, 256, 0, stream>>>(xb, wvb, nullptr, vtmp, M, N_EMBD, N_EMBD);

  // 4) V transpose first (vtmp read), then RoPE (qtmp freed for reuse as yb)
  vtrans_k<<<(unsigned)((nX + 255) / 256), 256, 0, stream>>>(vtmp, vtb);
  int nR = BATCH * SEQ * N_HEAD * 64;
  rope_regroup_k<<<(nR + 255) / 256, 256, 0, stream>>>(qtmp, qhb, c2, s2);
  rope_regroup_k<<<(nR + 255) / 256, 256, 0, stream>>>(ktmp, khb, c2, s2);

  // 5) causal flash attention -> y[B,T,C] bf16
  dim3 fgrid(SEQ / 64, BATCH * N_HEAD);
  flash_attn_k<<<fgrid, 128, 0, stream>>>(qhb, khb, vtb, yb);

  // 6) output projection -> fp32 d_out
  gemm_bt_k<true><<<ggrid, 256, 0, stream>>>(yb, wob, out, nullptr, M, N_EMBD, N_EMBD);

  (void)in_sizes; (void)n_in; (void)out_size; (void)ws_size;
}
